// RecognitionODERNN_22857815949632
// MI455X (gfx1250) — matrix operand order repaired and checked by reading, not compiled, not run on hardware
//
#include <hip/hip_runtime.h>
#include <hip/hip_bf16.h>
#include <math.h>

#define NH   4096
#define OBS  512
#define LAT  64
#define TPT  64      // time points -> 63 RK4 steps
#define NWG  128
#define NTHR 256

typedef __bf16 bf16_t;
typedef bf16_t v8bf  __attribute__((ext_vector_type(8)));
typedef bf16_t v16bf __attribute__((ext_vector_type(16)));
typedef float  v8f   __attribute__((ext_vector_type(8)));

union AFrag { v16bf v; v8bf h[2]; };

// ---------------------------------------------------------------------------
// split fp32 -> (hi, lo) bf16 pair; w ~= hi + lo with ~2^-16 relative error
// ---------------------------------------------------------------------------
__device__ __forceinline__ void bsplit(float v, bf16_t* hi, bf16_t* lo, int idx) {
    bf16_t h = (bf16_t)v;
    hi[idx] = h;
    lo[idx] = (bf16_t)(v - (float)h);
}

__global__ void split_bf16_kernel(const float* __restrict__ W,
                                  bf16_t* __restrict__ hi,
                                  bf16_t* __restrict__ lo,
                                  int n, unsigned* cnt) {
    if (blockIdx.x == 0 && threadIdx.x == 0) *cnt = 0u;   // reset grid barrier
    int i = blockIdx.x * blockDim.x + threadIdx.x;
    if (i < n) {
        float w = W[i];
        bf16_t h = (bf16_t)w;
        hi[i] = h;
        lo[i] = (bf16_t)(w - (float)h);
    }
}

// ---------------------------------------------------------------------------
// device-wide sense barrier (all NWG blocks resident; persistent kernel)
// ---------------------------------------------------------------------------
__device__ __forceinline__ void grid_sync(unsigned* cnt, unsigned& ep) {
    __threadfence();          // make this thread's global stores visible (agent)
    __syncthreads();
    ++ep;
    if (threadIdx.x == 0) {
        __hip_atomic_fetch_add(cnt, 1u, __ATOMIC_ACQ_REL, __HIP_MEMORY_SCOPE_AGENT);
        unsigned target = ep * (unsigned)NWG;
        while (__hip_atomic_load(cnt, __ATOMIC_ACQUIRE, __HIP_MEMORY_SCOPE_AGENT) < target)
            __builtin_amdgcn_s_sleep(1);
    }
    __syncthreads();
    __threadfence();          // invalidate before re-reading other WGs' data
}

// ---------------------------------------------------------------------------
// One wave computes a 16-row x [k0, k0+32*nchunk) slice of z = W*y using
// split-bf16 WMMA:  acc += Whi*yhi + Whi*ylo + Wlo*yhi  (fp32 accumulate).
// B operand = y broadcast across all 16 columns -> every column of D equals z.
// Partial sums combined across K-slice waves via LDS float atomics.
// unroll_count(4): bounded unroll so LLVM cannot hoist the whole A-stream
// into (extended) VGPRs and spill to scratch (seen at full unroll).
// ---------------------------------------------------------------------------
__device__ __forceinline__ void wmma_slice(const bf16_t* __restrict__ Whi,
                                           const bf16_t* __restrict__ Wlo,
                                           const bf16_t* __restrict__ yhi,
                                           const bf16_t* __restrict__ ylo,
                                           int K, int row0, int k0, int nchunk,
                                           float* zsum, int relbase) {
    const int lane  = threadIdx.x & 31;
    const int mlane = lane & 15;
    const int ahalf = (lane < 16) ? 0 : 8;    // doc'd 16-bit A layout
    const int bhalf = (lane < 16) ? 0 : 8;    // B matches A: low lanes K0-7,16-23; high K8-15,24-31

    const bf16_t* pah = Whi + (size_t)(row0 + mlane) * K + k0 + ahalf;
    const bf16_t* pal = Wlo + (size_t)(row0 + mlane) * K + k0 + ahalf;
    const bf16_t* pbh = yhi + k0 + bhalf;
    const bf16_t* pbl = ylo + k0 + bhalf;

    v8f acc = {};
#pragma clang loop unroll_count(4)
    for (int c = 0; c < nchunk; ++c) {
        AFrag ah, al;
        ah.h[0] = *(const v8bf*)(pah);        // K 0-7   (hi lanes: 8-15)
        ah.h[1] = *(const v8bf*)(pah + 16);   // K 16-23 (hi lanes: 24-31)
        al.h[0] = *(const v8bf*)(pal);
        al.h[1] = *(const v8bf*)(pal + 16);
        AFrag bhf, blf;
        bhf.h[0] = *(const v8bf*)(pbh);       // y[k..k+8) / y[k+8..k+16)
        bhf.h[1] = *(const v8bf*)(pbh + 16);  // y[k+16..k+24) / y[k+24..k+32)
        blf.h[0] = *(const v8bf*)(pbl);
        blf.h[1] = *(const v8bf*)(pbl + 16);
        v16bf bh = bhf.v;
        v16bf bl = blf.v;

        acc = __builtin_amdgcn_wmma_f32_16x16x32_bf16(false, ah.v, false, bh,
                                                      (short)0, acc, false, false);
        acc = __builtin_amdgcn_wmma_f32_16x16x32_bf16(false, ah.v, false, bl,
                                                      (short)0, acc, false, false);
        acc = __builtin_amdgcn_wmma_f32_16x16x32_bf16(false, al.v, false, bh,
                                                      (short)0, acc, false, false);
        pah += 32; pal += 32; pbh += 32; pbl += 32;
    }
    // D layout: lane 0 VGPR r = (M=r, N=0); lane 16 VGPR r = (M=8+r, N=0)
#pragma unroll
    for (int r = 0; r < 8; ++r) {
        float v = acc[r];
        if (lane == 0)  atomicAdd(&zsum[relbase + r], v);
        if (lane == 16) atomicAdd(&zsum[relbase + 8 + r], v);
    }
}

// ---------------------------------------------------------------------------
// Persistent kernel: 252 chained WMMA mat-vecs + RK4 updates + i2h + h2o
// ---------------------------------------------------------------------------
__global__ __launch_bounds__(NTHR, 1)
void odernn_persistent(const float* __restrict__ x, const float* __restrict__ h0,
                       const float* __restrict__ t,  const float* __restrict__ bf,
                       const float* __restrict__ Wi2h, const float* __restrict__ bi2h,
                       const float* __restrict__ Wh2o, const float* __restrict__ bh2o,
                       const bf16_t* __restrict__ Whi, const bf16_t* __restrict__ Wlo,
                       bf16_t* ybf_hi, bf16_t* ybf_lo,   // [2][NH] double-buffered
                       float* cvec, float* hvec,
                       unsigned* cnt, float* out) {
    __shared__ float zsum[32];

    const int tid  = threadIdx.x;
    const int g    = blockIdx.x;
    const int lane = tid & 31;
    const int wavei = __builtin_amdgcn_readfirstlane(tid >> 5); // scalar wave id
    const int tl  = wavei >> 2;            // local tile (0/1)
    const int kq  = wavei & 3;             // K quarter
    const int row0 = g * 32 + tl * 16;     // absolute 16-row tile base
    const int relbase = tl * 16;
    unsigned ep = 0;

    // wave 0 owns rows [g*32, g*32+32) for the elementwise RK4 state
    const int myrow = g * 32 + lane;
    float hreg = 0.f, accreg = 0.f;
    if (wavei == 0) {
        hreg = h0[myrow];
        bsplit(hreg, ybf_hi, ybf_lo, myrow);   // publish y = h into buffer 0
    }
    grid_sync(cnt, ep);

    int p = 0;                                  // y ping-pong parity
    for (int step = 0; step < TPT - 1; ++step) {
        float dt = t[step + 1] - t[step];
        for (int st = 0; st < 4; ++st) {
            if (wavei == 0) zsum[lane] = 0.f;
            __syncthreads();
            wmma_slice(Whi, Wlo, ybf_hi + p * NH, ybf_lo + p * NH,
                       NH, row0, kq * 1024, 32, zsum, relbase);
            __syncthreads();
            if (wavei == 0) {
                float k = tanhf(zsum[lane] + bf[myrow]);
                float ynext;
                if      (st == 0) { accreg  = k;        ynext = hreg + 0.5f * dt * k; }
                else if (st == 1) { accreg += 2.f * k;  ynext = hreg + 0.5f * dt * k; }
                else if (st == 2) { accreg += 2.f * k;  ynext = hreg + dt * k; }
                else              { accreg += k;
                                    hreg   += (dt * (1.f / 6.f)) * accreg;
                                    ynext   = hreg; }
                bsplit(ynext, ybf_hi + (1 - p) * NH, ybf_lo + (1 - p) * NH, myrow);
            }
            grid_sync(cnt, ep);
            p ^= 1;
        }
    }

    // combined = [x(512), hT(4096)] in fp32 for one-shot i2h
    if (wavei == 0) {
        cvec[OBS + myrow] = hreg;
        if (g < OBS / 32) cvec[g * 32 + lane] = x[g * 32 + lane];
    }
    grid_sync(cnt, ep);

    // i2h: 4096 rows x 4608, fp32 dot per wave-row (single pass, bw-trivial)
    {
        const int wid = g * 8 + wavei;         // 0..1023
        for (int j = 0; j < 4; ++j) {
            const int row = wid * 4 + j;
            const float4* wp = (const float4*)(Wi2h + (size_t)row * (OBS + NH));
            const float4* cp = (const float4*)cvec;
            float s = 0.f;
            for (int i = lane; i < (OBS + NH) / 4; i += 32) {
                float4 a = wp[i], b = cp[i];
                s += a.x * b.x + a.y * b.y + a.z * b.z + a.w * b.w;
            }
            for (int off = 16; off; off >>= 1) s += __shfl_xor(s, off, 32);
            if (lane == 0) {
                float hn = tanhf(s + bi2h[row]);
                out[128 + row] = hn;           // h_new output
                hvec[row] = hn;
            }
        }
    }
    grid_sync(cnt, ep);

    // h2o: 128 rows x 4096 on WGs 0..3
    if (g < 4) {
        const int wid = g * 8 + wavei;         // 0..31
        for (int j = 0; j < 4; ++j) {
            const int row = wid * 4 + j;       // 0..127
            const float4* wp = (const float4*)(Wh2o + (size_t)row * NH);
            const float4* hp = (const float4*)hvec;
            float s = 0.f;
            for (int i = lane; i < NH / 4; i += 32) {
                float4 a = wp[i], b = hp[i];
                s += a.x * b.x + a.y * b.y + a.z * b.z + a.w * b.w;
            }
            for (int off = 16; off; off >>= 1) s += __shfl_xor(s, off, 32);
            if (lane == 0) out[row] = s + bh2o[row];   // out output (no tanh)
        }
    }
}

// ---------------------------------------------------------------------------
extern "C" void kernel_launch(void* const* d_in, const int* in_sizes, int n_in,
                              void* d_out, int out_size, void* d_ws, size_t ws_size,
                              hipStream_t stream) {
    const float* x    = (const float*)d_in[0];
    const float* h0   = (const float*)d_in[1];
    const float* t    = (const float*)d_in[2];
    const float* Wf   = (const float*)d_in[3];
    const float* bf   = (const float*)d_in[4];
    const float* Wi2h = (const float*)d_in[5];
    const float* bi2h = (const float*)d_in[6];
    const float* Wh2o = (const float*)d_in[7];
    const float* bh2o = (const float*)d_in[8];
    float* out = (float*)d_out;

    char* ws = (char*)d_ws;
    size_t off = 0;
    auto take = [&](size_t bytes) {
        char* p = ws + off;
        off = (off + bytes + 255) & ~(size_t)255;
        return p;
    };
    unsigned* cnt = (unsigned*)take(256);
    bf16_t* Whi = (bf16_t*)take((size_t)NH * NH * 2);
    bf16_t* Wlo = (bf16_t*)take((size_t)NH * NH * 2);
    bf16_t* yhi = (bf16_t*)take((size_t)2 * NH * 2);
    bf16_t* ylo = (bf16_t*)take((size_t)2 * NH * 2);
    float*  cvec = (float*)take((size_t)(OBS + NH) * 4);
    float*  hvec = (float*)take((size_t)NH * 4);
    (void)ws_size; (void)in_sizes; (void)n_in; (void)out_size;

    const int n = NH * NH;
    split_bf16_kernel<<<(n + 255) / 256, 256, 0, stream>>>(Wf, Whi, Wlo, n, cnt);
    odernn_persistent<<<NWG, NTHR, 0, stream>>>(x, h0, t, bf, Wi2h, bi2h, Wh2o, bh2o,
                                                Whi, Wlo, yhi, ylo, cvec, hvec, cnt, out);
}
